// CIFAR10BinaryNet_86311662780727
// MI455X (gfx1250) — 8-bit matrix operand order repaired and checked by reading, compiled, not run on hardware
//
#include <hip/hip_runtime.h>
#include <stdint.h>

typedef __attribute__((ext_vector_type(8))) int v8i;

#define BN_EPS 1e-4f
#define NB 256

__device__ __forceinline__ float binf(float w) {
    float t = fminf(fmaxf((w + 1.0f) * 0.5f, 0.0f), 1.0f);
    return 2.0f * rintf(t) - 1.0f;
}
__device__ __forceinline__ int8_t bin8(float w) { return (int8_t)binf(w); }

__device__ __forceinline__ void async_copy16(void* lds, const void* g) {
    asm volatile("global_load_async_to_lds_b128 %0, %1, off"
                 :: "v"((uint32_t)(uintptr_t)lds), "v"(g) : "memory");
}
__device__ __forceinline__ void wait_async() {
    asm volatile("s_wait_asynccnt 0x0" ::: "memory");
}

__global__ void k_binw_conv(const float* __restrict__ src, int8_t* __restrict__ dst,
                            int Cin, int n) {
    int idx = blockIdx.x * 256 + threadIdx.x;
    if (idx >= n) return;
    int K   = Cin * 9;
    int co  = idx / K;
    int r   = idx - co * K;
    int kyx = r / Cin;
    int ci  = r - kyx * Cin;
    dst[idx] = bin8(src[(size_t)(co * Cin + ci) * 9 + kyx]);
}

__global__ void k_binw_fc1(const float* __restrict__ src, int8_t* __restrict__ dst) {
    int idx = blockIdx.x * 256 + threadIdx.x;
    if (idx >= 1024 * 8192) return;
    int o  = idx >> 13;
    int r  = idx & 8191;
    int c  = r >> 4;
    int hw = r & 15;
    dst[((size_t)o << 13) + hw * 512 + c] = bin8(src[idx]);
}

__global__ void k_binw_copy(const float* __restrict__ src, int8_t* __restrict__ dst, int n) {
    int idx = blockIdx.x * 256 + threadIdx.x;
    if (idx < n) dst[idx] = bin8(src[idx]);
}

__global__ void k_zero(float* __restrict__ p, int n) {
    int idx = blockIdx.x * 256 + threadIdx.x;
    if (idx < n) p[idx] = 0.0f;
}

__global__ void k_conv1(const float* __restrict__ x, const float* __restrict__ w,
                        const float* __restrict__ cb, float* __restrict__ out) {
    int idx = blockIdx.x * 256 + threadIdx.x;
    if (idx >= NB * 32 * 32 * 128) return;
    int c = idx & 127;
    int p = idx >> 7;
    int ox = p & 31; p >>= 5;
    int oy = p & 31; p >>= 5;
    int b  = p;
    float s = cb[c];
    #pragma unroll
    for (int ci = 0; ci < 3; ++ci)
        #pragma unroll
        for (int ky = 0; ky < 3; ++ky) {
            int iy = oy + ky - 1;
            if (iy < 0 || iy >= 32) continue;
            #pragma unroll
            for (int kx = 0; kx < 3; ++kx) {
                int ix = ox + kx - 1;
                if (ix < 0 || ix >= 32) continue;
                float xv = x[(((size_t)b * 3 + ci) * 32 + iy) * 32 + ix];
                float wv = binf(w[(((size_t)c * 3 + ci) * 3 + ky) * 3 + kx]);
                s += xv * wv;
            }
        }
    out[idx] = s;
}

template<int CIN, int COUT, int H, int W>
__global__ __launch_bounds__(256)
void k_conv_wmma(const int8_t* __restrict__ act, const int8_t* __restrict__ wts,
                 const float* __restrict__ bias, float* __restrict__ out) {
    constexpr int K = CIN * 9;
    constexpr int S = H * W;
    __shared__ __align__(16) int8_t lA[64 * 64];
    __shared__ __align__(16) int8_t lB[128 * 64];
    const int tid  = threadIdx.x;
    const int lane = tid & 31;
    const int wave = tid >> 5;
    const int wm = wave & 3;
    const int wn = wave >> 2;
    const int m0 = blockIdx.y * 64;
    const int n0 = blockIdx.x * 128;

    const int am = tid >> 2;
    const int ak = (tid & 3) * 16;
    const int gp  = m0 + am;
    const int img = gp / S;
    const int pix = gp - img * S;
    const int oy  = pix / W;
    const int ox  = pix - oy * W;
    const int bn  = tid >> 1;
    const int bkh = (tid & 1) * 32;

    v8i acc[4] = {};

    for (int kyx = 0; kyx < 9; ++kyx) {
        const int ky = kyx / 3, kx = kyx - 3 * (kyx / 3);
        const int iy = oy + ky - 1;
        const int ix = ox + kx - 1;
        const bool inb = (iy >= 0) & (iy < H) & (ix >= 0) & (ix < W);
        const int8_t* arow =
            act + (((size_t)img * H + iy) * W + ix) * CIN + ak;
        const int8_t* wrow = wts + (size_t)(n0 + bn) * K + kyx * CIN + bkh;

        for (int ci0 = 0; ci0 < CIN; ci0 += 64) {
            if (inb) {
                async_copy16(lA + am * 64 + ak, arow + ci0);
            } else {
                *(int4*)(lA + am * 64 + ak) = make_int4(0, 0, 0, 0);
            }
            async_copy16(lB + bn * 64 + bkh,      wrow + ci0);
            async_copy16(lB + bn * 64 + bkh + 16, wrow + ci0 + 16);
            if (inb && ci0 + 64 < CIN)
                __builtin_prefetch(arow + ci0 + 64, 0, 1);
            wait_async();
            __syncthreads();

            v8i a;
            {
                const int8_t* ap = lA + (wm * 16 + (lane & 15)) * 64 + (lane >> 4) * 16;
                #pragma unroll
                for (int j = 0; j < 4; ++j) {
                    const int2 d = *(const int2*)(ap + (j & 1) * 8 + (j >> 1) * 32);
                    a[2 * j] = d.x; a[2 * j + 1] = d.y;
                }
            }
            #pragma unroll
            for (int s = 0; s < 4; ++s) {
                const int8_t* bp =
                    lB + (wn * 64 + s * 16 + (lane & 15)) * 64 + (lane >> 4) * 16;
                const int4 lo = *(const int4*)(bp);
                const int4 hi = *(const int4*)(bp + 32);
                v8i b;
                b[0] = lo.x; b[1] = lo.y; b[2] = lo.z; b[3] = lo.w;
                b[4] = hi.x; b[5] = hi.y; b[6] = hi.z; b[7] = hi.w;
                acc[s] = __builtin_amdgcn_wmma_i32_16x16x64_iu8(
                    true, a, true, b, acc[s], false, false);
            }
            __syncthreads();
        }
    }

    #pragma unroll
    for (int s = 0; s < 4; ++s) {
        const int ch = n0 + wn * 64 + s * 16 + (lane & 15);
        const float bv = bias[ch];
        #pragma unroll
        for (int r = 0; r < 8; ++r) {
            const int p = m0 + wm * 16 + (lane >> 4) * 8 + r;
            out[(size_t)p * COUT + ch] = (float)acc[s][r] + bv;
        }
    }
}

__global__ __launch_bounds__(256)
void k_fc_wmma(const int8_t* __restrict__ act, const int8_t* __restrict__ wts,
               const float* __restrict__ bias, float* __restrict__ out,
               int K, int Cout) {
    __shared__ __align__(16) int8_t lA[64 * 64];
    __shared__ __align__(16) int8_t lB[128 * 64];
    const int tid  = threadIdx.x;
    const int lane = tid & 31;
    const int wave = tid >> 5;
    const int wm = wave & 3;
    const int wn = wave >> 2;
    const int m0 = blockIdx.y * 64;
    const int n0 = blockIdx.x * 128;
    const int am = tid >> 2;
    const int ak = (tid & 3) * 16;
    const int bn  = tid >> 1;
    const int bkh = (tid & 1) * 32;

    v8i acc[4] = {};

    const int8_t* arow = act + (size_t)(m0 + am) * K + ak;
    const int8_t* wrow = wts + (size_t)(n0 + bn) * K + bkh;

    for (int kt = 0; kt < K; kt += 64) {
        async_copy16(lA + am * 64 + ak, arow + kt);
        async_copy16(lB + bn * 64 + bkh,      wrow + kt);
        async_copy16(lB + bn * 64 + bkh + 16, wrow + kt + 16);
        if (kt + 64 < K) __builtin_prefetch(wrow + kt + 64, 0, 1);
        wait_async();
        __syncthreads();

        v8i a;
        {
            const int8_t* ap = lA + (wm * 16 + (lane & 15)) * 64 + (lane >> 4) * 16;
            #pragma unroll
            for (int j = 0; j < 4; ++j) {
                const int2 d = *(const int2*)(ap + (j & 1) * 8 + (j >> 1) * 32);
                a[2 * j] = d.x; a[2 * j + 1] = d.y;
            }
        }
        #pragma unroll
        for (int s = 0; s < 4; ++s) {
            const int8_t* bp =
                lB + (wn * 64 + s * 16 + (lane & 15)) * 64 + (lane >> 4) * 16;
            const int4 lo = *(const int4*)(bp);
            const int4 hi = *(const int4*)(bp + 32);
            v8i b;
            b[0] = lo.x; b[1] = lo.y; b[2] = lo.z; b[3] = lo.w;
            b[4] = hi.x; b[5] = hi.y; b[6] = hi.z; b[7] = hi.w;
            acc[s] = __builtin_amdgcn_wmma_i32_16x16x64_iu8(
                true, a, true, b, acc[s], false, false);
        }
        __syncthreads();
    }

    #pragma unroll
    for (int s = 0; s < 4; ++s) {
        const int ch = n0 + wn * 64 + s * 16 + (lane & 15);
        const float bv = bias[ch];
        #pragma unroll
        for (int r = 0; r < 8; ++r) {
            const int row = m0 + wm * 16 + (lane >> 4) * 8 + r;
            out[(size_t)row * Cout + ch] = (float)acc[s][r] + bv;
        }
    }
}

__global__ void k_pool(const float* __restrict__ in, float* __restrict__ out,
                       int H, int W, int C) {
    int Ho = H >> 1, Wo = W >> 1;
    int n = NB * Ho * Wo * C;
    int idx = blockIdx.x * 256 + threadIdx.x;
    if (idx >= n) return;
    int c = idx % C;
    int t = idx / C;
    int xo = t % Wo; t /= Wo;
    int yo = t % Ho; t /= Ho;
    int b  = t;
    size_t base = (((size_t)b * H + 2 * yo) * W + 2 * xo) * C + c;
    float m = in[base];
    m = fmaxf(m, in[base + C]);
    m = fmaxf(m, in[base + (size_t)W * C]);
    m = fmaxf(m, in[base + (size_t)W * C + C]);
    out[idx] = m;
}

__global__ void k_stats(const float* __restrict__ y, float* __restrict__ stats,
                        int R, int C) {
    __shared__ float ls[256], lq[256];
    int tid = threadIdx.x;
    int c = blockIdx.x * 64 + (tid & 63);
    int g = tid >> 6;
    int rows_per = (R + gridDim.y - 1) / gridDim.y;
    int r0 = blockIdx.y * rows_per;
    int r1 = min(r0 + rows_per, R);
    float s = 0.0f, q = 0.0f;
    if (c < C)
        for (int r = r0 + g; r < r1; r += 4) {
            float v = y[(size_t)r * C + c];
            s += v; q += v * v;
        }
    ls[tid] = s; lq[tid] = q;
    __syncthreads();
    if (g == 0 && c < C) {
        #pragma unroll
        for (int j = 1; j < 4; ++j) { s += ls[tid + 64 * j]; q += lq[tid + 64 * j]; }
        atomicAdd(&stats[c], s);
        atomicAdd(&stats[C + c], q);
    }
}

__global__ void k_bn_btanh_i8(const float* __restrict__ y, const float* __restrict__ stats,
                              const float* __restrict__ gam, const float* __restrict__ bet,
                              const float* __restrict__ u, int8_t* __restrict__ out,
                              int R, int C, int S) {
    int idx = blockIdx.x * 256 + threadIdx.x;
    if (idx >= R * C) return;
    int c = idx % C;
    int r = idx / C;
    int b = r / S;
    int s = r - b * S;
    float invR = 1.0f / (float)R;
    float m = stats[c] * invR;
    float v = stats[C + c] * invR - m * m;
    float inv = gam[c] * rsqrtf(v + BN_EPS);
    float f = (y[idx] - m) * inv + bet[c];
    float p = fminf(fmaxf((f + 1.0f) * 0.5f, 0.0f), 1.0f);
    float uu = u[((size_t)b * C + c) * S + s];
    out[idx] = (uu < p) ? (int8_t)1 : (int8_t)-1;
}

__global__ void k_bn_final(const float* __restrict__ y, const float* __restrict__ stats,
                           const float* __restrict__ gam, const float* __restrict__ bet,
                           float* __restrict__ out, int R, int C) {
    int idx = blockIdx.x * 256 + threadIdx.x;
    if (idx >= R * C) return;
    int c = idx % C;
    float invR = 1.0f / (float)R;
    float m = stats[c] * invR;
    float v = stats[C + c] * invR - m * m;
    float inv = gam[c] * rsqrtf(v + BN_EPS);
    out[idx] = (y[idx] - m) * inv + bet[c];
}

__global__ void k_fc3(const int8_t* __restrict__ act, const int8_t* __restrict__ w,
                      const float* __restrict__ fb, float* __restrict__ out) {
    int idx = blockIdx.x * 256 + threadIdx.x;
    if (idx >= NB * 10) return;
    int o = idx % 10;
    int b = idx / 10;
    int acc = 0;
    const int8_t* a = act + (size_t)b * 1024;
    const int8_t* ww = w + (size_t)o * 1024;
    for (int k = 0; k < 1024; ++k) acc += (int)a[k] * (int)ww[k];
    out[idx] = (float)acc + fb[o];
}

extern "C" void kernel_launch(void* const* d_in, const int* in_sizes, int n_in,
                              void* d_out, int out_size, void* d_ws, size_t ws_size,
                              hipStream_t stream) {
    (void)in_sizes; (void)n_in; (void)out_size; (void)ws_size;
    const float* x   = (const float*)d_in[0];
    const float* w[7];   const float* cb[7];  const float* g[7];  const float* be[7];
    for (int i = 1; i <= 6; ++i) {
        w[i]  = (const float*)d_in[1 + (i - 1) * 4];
        cb[i] = (const float*)d_in[2 + (i - 1) * 4];
        g[i]  = (const float*)d_in[3 + (i - 1) * 4];
        be[i] = (const float*)d_in[4 + (i - 1) * 4];
    }
    const float* fw[4]; const float* fb[4]; const float* fg[4]; const float* fbe[4];
    for (int i = 1; i <= 3; ++i) {
        fw[i]  = (const float*)d_in[25 + (i - 1) * 4];
        fb[i]  = (const float*)d_in[26 + (i - 1) * 4];
        fg[i]  = (const float*)d_in[27 + (i - 1) * 4];
        fbe[i] = (const float*)d_in[28 + (i - 1) * 4];
    }
    const float* u[9];
    for (int i = 1; i <= 8; ++i) u[i] = (const float*)d_in[36 + i];

    uint8_t* ws = (uint8_t*)d_ws;
    size_t off = 0;
    auto take = [&](size_t bytes) -> void* {
        void* p = ws + off;
        off = (off + bytes + 255) & ~(size_t)255;
        return p;
    };
    int8_t* wc2 = (int8_t*)take(128 * 1152);
    int8_t* wc3 = (int8_t*)take(256 * 1152);
    int8_t* wc4 = (int8_t*)take(256 * 2304);
    int8_t* wc5 = (int8_t*)take(512 * 2304);
    int8_t* wc6 = (int8_t*)take(512 * 4608);
    int8_t* wf1 = (int8_t*)take((size_t)1024 * 8192);
    int8_t* wf2 = (int8_t*)take((size_t)1024 * 1024);
    int8_t* wf3 = (int8_t*)take(10 * 1024);
    float*  ybuf  = (float*)take((size_t)NB * 32 * 32 * 128 * 4);
    float*  ypool = (float*)take((size_t)NB * 16 * 16 * 128 * 4);
    int8_t* actA  = (int8_t*)take((size_t)NB * 32 * 32 * 128);
    int8_t* actB  = (int8_t*)take((size_t)NB * 16 * 16 * 256);
    float*  stats = (float*)take(2 * 1024 * 4);
    float*  y3    = (float*)take(NB * 10 * 4);

    auto blocks = [](long long n) { return (unsigned)((n + 255) / 256); };

    k_binw_conv<<<blocks(128 * 1152), 256, 0, stream>>>(w[2], wc2, 128, 128 * 1152);
    k_binw_conv<<<blocks(256 * 1152), 256, 0, stream>>>(w[3], wc3, 128, 256 * 1152);
    k_binw_conv<<<blocks(256 * 2304), 256, 0, stream>>>(w[4], wc4, 256, 256 * 2304);
    k_binw_conv<<<blocks(512 * 2304), 256, 0, stream>>>(w[5], wc5, 256, 512 * 2304);
    k_binw_conv<<<blocks(512 * 4608), 256, 0, stream>>>(w[6], wc6, 512, 512 * 4608);
    k_binw_fc1<<<blocks((long long)1024 * 8192), 256, 0, stream>>>(fw[1], wf1);
    k_binw_copy<<<blocks(1024 * 1024), 256, 0, stream>>>(fw[2], wf2, 1024 * 1024);
    k_binw_copy<<<blocks(10 * 1024), 256, 0, stream>>>(fw[3], wf3, 10 * 1024);

    auto bn_stage = [&](const float* y, int R, int C, int S, const float* gg,
                        const float* bb, const float* uu, int8_t* aout) {
        int chunks = (R >= 16384) ? 64 : 1;
        k_zero<<<blocks(2 * C), 256, 0, stream>>>(stats, 2 * C);
        k_stats<<<dim3((C + 63) / 64, chunks), 256, 0, stream>>>(y, stats, R, C);
        k_bn_btanh_i8<<<blocks((long long)R * C), 256, 0, stream>>>(
            y, stats, gg, bb, uu, aout, R, C, S);
    };

    k_conv1<<<blocks((long long)NB * 32 * 32 * 128), 256, 0, stream>>>(x, w[1], cb[1], ybuf);
    bn_stage(ybuf, NB * 1024, 128, 1024, g[1], be[1], u[1], actA);

    k_conv_wmma<128, 128, 32, 32><<<dim3(1, NB * 1024 / 64), 256, 0, stream>>>(
        actA, wc2, cb[2], ybuf);
    k_pool<<<blocks((long long)NB * 256 * 128), 256, 0, stream>>>(ybuf, ypool, 32, 32, 128);
    bn_stage(ypool, NB * 256, 128, 256, g[2], be[2], u[2], actB);

    k_conv_wmma<128, 256, 16, 16><<<dim3(2, NB * 256 / 64), 256, 0, stream>>>(
        actB, wc3, cb[3], ybuf);
    bn_stage(ybuf, NB * 256, 256, 256, g[3], be[3], u[3], actA);

    k_conv_wmma<256, 256, 16, 16><<<dim3(2, NB * 256 / 64), 256, 0, stream>>>(
        actA, wc4, cb[4], ybuf);
    k_pool<<<blocks((long long)NB * 64 * 256), 256, 0, stream>>>(ybuf, ypool, 16, 16, 256);
    bn_stage(ypool, NB * 64, 256, 64, g[4], be[4], u[4], actB);

    k_conv_wmma<256, 512, 8, 8><<<dim3(4, NB * 64 / 64), 256, 0, stream>>>(
        actB, wc5, cb[5], ybuf);
    bn_stage(ybuf, NB * 64, 512, 64, g[5], be[5], u[5], actA);

    k_conv_wmma<512, 512, 8, 8><<<dim3(4, NB * 64 / 64), 256, 0, stream>>>(
        actA, wc6, cb[6], ybuf);
    k_pool<<<blocks((long long)NB * 16 * 512), 256, 0, stream>>>(ybuf, ypool, 8, 8, 512);
    bn_stage(ypool, NB * 16, 512, 16, g[6], be[6], u[6], actB);

    k_fc_wmma<<<dim3(1024 / 128, NB / 64), 256, 0, stream>>>(actB, wf1, fb[1], ybuf, 8192, 1024);
    bn_stage(ybuf, NB, 1024, 1, fg[1], fbe[1], u[7], actA);

    k_fc_wmma<<<dim3(1024 / 128, NB / 64), 256, 0, stream>>>(actA, wf2, fb[2], ybuf, 1024, 1024);
    bn_stage(ybuf, NB, 1024, 1, fg[2], fbe[2], u[8], actB);

    k_fc3<<<blocks(NB * 10), 256, 0, stream>>>(actB, wf3, fb[3], y3);
    k_zero<<<1, 256, 0, stream>>>(stats, 20);
    k_stats<<<dim3(1, 1), 256, 0, stream>>>(y3, stats, NB, 10);
    k_bn_final<<<blocks(NB * 10), 256, 0, stream>>>(y3, stats, fg[3], fbe[3],
                                                    (float*)d_out, NB, 10);
}
